// DPFPAttention_41180146434409
// MI455X (gfx1250) — matrix operand order repaired and checked by reading, not compiled, not run on hardware
//
#include <hip/hip_runtime.h>

// ---------------------------------------------------------------------------
// DPFP delta-rule attention layer for MI455X (gfx1250, wave32, WMMA bf16,
// async global->LDS staging, double-buffered, branch-free pipelined loop)
// ---------------------------------------------------------------------------

#define DIM       1024
#define HEADS     16
#define DIM_HEAD  64
#define FDIM      128            // 2*NU*DIM_HEAD
#define BATCH     4
#define SEQ       4096
#define ROWS      (BATCH * SEQ)  // 16384
#define BH        (BATCH * HEADS)
#define SCALE_F   (0.125f)       // 1/sqrt(64)
#define LN_EPS    (1e-5f)

typedef __attribute__((ext_vector_type(16))) __bf16 v16bf;
typedef __attribute__((ext_vector_type(8)))  __bf16 v8bf;
typedef __attribute__((ext_vector_type(8)))  float  v8f;

__device__ __forceinline__ __bf16 f2bf(float f) {
    unsigned u = __builtin_bit_cast(unsigned, f);
    unsigned r = (u + 0x7FFFu + ((u >> 16) & 1u)) >> 16;
    unsigned short s = (unsigned short)r;
    return __builtin_bit_cast(__bf16, s);
}

// gfx1250 async copy: 16B per lane, global -> LDS, tracked by ASYNCcnt.
__device__ __forceinline__ void async_b128(__bf16* lds, const __bf16* g) {
    unsigned loff = (unsigned)(unsigned long long)lds;   // low 32 bits = LDS byte addr
    asm volatile("global_load_async_to_lds_b128 %0, %1, off"
                 :: "v"(loff), "v"(g) : "memory");
}
__device__ __forceinline__ void wait_async_le3() {
    asm volatile("s_wait_asynccnt 0x3" ::: "memory");
}
__device__ __forceinline__ void wait_async_0() {
    asm volatile("s_wait_asynccnt 0x0" ::: "memory");
}

// ---------------------------------------------------------------------------
// Generic LDS-tiled bf16 WMMA GEMM: C[128 x 64] tile per 256-thread block.
// A row-major [M,K] lda ; BT row-major [N,K] ldbT (i.e. B column-major).
// Both stage identically as contiguous-K slabs via async LDS copies,
// double-buffered (3 async ops / wave / tile), 2x-unrolled pipelined loop.
//   C = alpha * (A @ B) [+ cinit] [+ bias(col)] [+ resid(elem)]
// Requires K % 64 == 0 (nsteps even, >= 2).
// ---------------------------------------------------------------------------
__device__ __forceinline__ void gemm_tile_bf16(
    const __bf16* __restrict__ A,  int lda,
    const __bf16* __restrict__ BT, int ldbT,
    int K, int tmBase, int nBase, float alpha,
    const float* __restrict__ cinit, int ldci,
    const float* __restrict__ bias,
    const float* __restrict__ resid,
    float* __restrict__ cf, __bf16* __restrict__ cbf,
    __bf16* __restrict__ cbfT, int ldcT, int ldc)
{
    __shared__ __align__(64) __bf16 ldsA[2][128 * 32];  // row-major [128][32]
    __shared__ __align__(64) __bf16 ldsB[2][64 * 32];   // BT row-major [64][32]

    const int tid  = threadIdx.x;
    const int lane = tid & 31;
    const int wave = tid >> 5;
    const int half = lane >> 4;     // 0: lanes 0-15, 1: lanes 16-31
    const int l16  = lane & 15;

    // chunk mapping (8 bf16 = 16B per async op)
    const int ar0 = tid >> 2,         akc0 = (tid & 3) << 3;          // A chunk 1
    const int ar1 = (tid + 256) >> 2, akc1 = ((tid + 256) & 3) << 3;  // A chunk 2
    const int br  = tid >> 2,         bkc  = (tid & 3) << 3;          // BT chunk

    const __bf16* gA0 = &A[(size_t)(tmBase + ar0) * lda + (size_t)akc0];
    const __bf16* gA1 = &A[(size_t)(tmBase + ar1) * lda + (size_t)akc1];
    const __bf16* gB  = &BT[(size_t)(nBase + br) * ldbT + (size_t)bkc];

    auto stage = [&](int buf, int k0) {
        async_b128(&ldsA[buf][ar0 * 32 + akc0], gA0 + k0);
        async_b128(&ldsA[buf][ar1 * 32 + akc1], gA1 + k0);
        async_b128(&ldsB[buf][br * 32 + bkc],   gB  + k0);
    };

    v8f acc[4];
#pragma unroll
    for (int i = 0; i < 4; ++i) acc[i] = (v8f){0.f,0.f,0.f,0.f,0.f,0.f,0.f,0.f};

    const int arow = wave * 16 + l16;
    auto compute = [&](int buf) {   // buf is a compile-time constant at call sites
        // A fragment (16x32): lane<16 holds K 0..15, lane>=16 K 16..31
        union { v16bf v; v8bf h[2]; } af;
        af.h[0] = *(const v8bf*)&ldsA[buf][arow * 32 + half * 16];
        af.h[1] = *(const v8bf*)&ldsA[buf][arow * 32 + half * 16 + 8];
#pragma unroll
        for (int nt = 0; nt < 4; ++nt) {
            // B fragment (32x16): lane -> col l16; lane<16 K 0..15, lane>=16 K 16..31
            v16bf bfrag = *(const v16bf*)&ldsB[buf][(nt * 16 + l16) * 32 + half * 16];
            acc[nt] = __builtin_amdgcn_wmma_f32_16x16x32_bf16(
                false, af.v, false, bfrag, (short)0, acc[nt], false, false);
        }
    };

    const int nsteps = K >> 5;      // even, >= 2
    stage(0, 0);
    int k = 0;
    for (; k + 2 < nsteps; k += 2) {
        stage(1, (k + 1) << 5); wait_async_le3();
        __syncthreads(); compute(0); __syncthreads();
        stage(0, (k + 2) << 5); wait_async_le3();
        __syncthreads(); compute(1); __syncthreads();
    }
    // tail: exactly 2 steps remain (k == nsteps-2)
    stage(1, (k + 1) << 5); wait_async_le3();
    __syncthreads(); compute(0); __syncthreads();
    wait_async_0();
    __syncthreads(); compute(1);

    // write back: v8f element j -> row wave*16 + half*8 + j, col nt*16 + l16
    const int rbase = wave * 16 + half * 8;
#pragma unroll
    for (int nt = 0; nt < 4; ++nt) {
        const int ncol = nBase + nt * 16 + l16;
#pragma unroll
        for (int j = 0; j < 8; ++j) {
            const int mrow = tmBase + rbase + j;
            float v = acc[nt][j] * alpha;
            if (cinit) v += cinit[(size_t)mrow * ldci + ncol];
            if (bias)  v += bias[ncol];
            if (resid) v += resid[(size_t)mrow * ldc + ncol];
            if (cf)   cf[(size_t)mrow * ldc + ncol]   = v;
            if (cbf)  cbf[(size_t)mrow * ldc + ncol]  = f2bf(v);
            if (cbfT) cbfT[(size_t)ncol * ldcT + mrow] = f2bf(v);
        }
    }
}

// ---------------------------------------------------------------------------
// Elementwise / helper kernels
// ---------------------------------------------------------------------------
__global__ void k_f32_to_bf16(const float* __restrict__ in, __bf16* __restrict__ out, long n) {
    long i = (long)blockIdx.x * blockDim.x + threadIdx.x;
    long stride = (long)gridDim.x * blockDim.x;
    for (; i < n; i += stride) out[i] = f2bf(in[i]);
}

// Transposing convert: in f32 [R][C] -> out bf16 [C][R]; 32x32 tiles, block (32,8)
__global__ void k_f32_to_bf16_T(const float* __restrict__ in, __bf16* __restrict__ out,
                                int R, int C) {
    __shared__ float t[32][33];
    const int tr = blockIdx.y * 32, tc = blockIdx.x * 32;
    for (int j = threadIdx.y; j < 32; j += 8)
        t[j][threadIdx.x] = in[(size_t)(tr + j) * C + tc + threadIdx.x];
    __syncthreads();
    for (int j = threadIdx.y; j < 32; j += 8)
        out[(size_t)(tc + j) * R + tr + threadIdx.x] = f2bf(t[threadIdx.x][j]);
}

// qkv = x @ Wqkv   [16384,1024]x[1024,3072] -> f32
__global__ void k_gemm_qkv(const __bf16* __restrict__ xbf, const __bf16* __restrict__ wqkvT,
                           float* __restrict__ qkv) {
    const int ntiles = 3 * DIM / 64;           // 48
    int tm = blockIdx.x / ntiles, tn = blockIdx.x % ntiles;
    gemm_tile_bf16(xbf, DIM, wqkvT, DIM, DIM, tm * 128, tn * 64, 1.0f,
                   nullptr, 0, nullptr, nullptr, qkv, nullptr, nullptr, 0, 3 * DIM);
}

// beta[bh][n] = x[row] . Wbeta[:,h] + bbeta[h]
__global__ void k_beta(const float* __restrict__ x, const float* __restrict__ Wbeta,
                       const float* __restrict__ bbeta, float* __restrict__ beta) {
    int idx = blockIdx.x * blockDim.x + threadIdx.x;    // 0..ROWS*HEADS-1
    int r = idx >> 4, h = idx & 15;
    float s = bbeta[h];
    const float* xr = x + (size_t)r * DIM;
    for (int c = 0; c < DIM; ++c) s += xr[c] * Wbeta[c * HEADS + h];
    int b = r >> 12, n = r & (SEQ - 1);
    beta[((size_t)(b * HEADS + h) << 12) + n] = s;
}

// dpfp feature map + sum-normalize + dv = beta*(v - W0^T k); 4 positions/block
__global__ void k_dpfp(const float* __restrict__ qkv, const float* __restrict__ beta,
                       const float* __restrict__ W0,
                       __bf16* __restrict__ qn, __bf16* __restrict__ knT,
                       __bf16* __restrict__ dvT) {
    __shared__ float aq[4][FDIM], ak[4][FDIM], pq[4][FDIM], pk[4][FDIM], knn[4][FDIM];
    __shared__ float rq[4][64], rk[4][64];
    const int tid = threadIdx.x, pos = tid >> 6, t = tid & 63;
    const long p = (long)blockIdx.x * 4 + pos;        // bh*SEQ + n
    const int bh = (int)(p >> 12), n = (int)(p & (SEQ - 1));
    const int b = bh >> 4, h = bh & 15;
    const size_t row = (size_t)b * SEQ + n;

    const float qv = qkv[row * (3 * DIM) + h * DIM_HEAD + t];
    const float kv = qkv[row * (3 * DIM) + DIM + h * DIM_HEAD + t];
    const float vv = qkv[row * (3 * DIM) + 2 * DIM + h * DIM_HEAD + t];

    aq[pos][t] = fmaxf(qv, 0.f);  aq[pos][t + 64] = fmaxf(-qv, 0.f);
    ak[pos][t] = fmaxf(kv, 0.f);  ak[pos][t + 64] = fmaxf(-kv, 0.f);
    __syncthreads();
#pragma unroll
    for (int ii = 0; ii < 2; ++ii) {   // dpfp: a[i] * a[(i-1) mod 128]
        int i = t + ii * 64;
        pq[pos][i] = aq[pos][i] * aq[pos][(i + FDIM - 1) & (FDIM - 1)];
        pk[pos][i] = ak[pos][i] * ak[pos][(i + FDIM - 1) & (FDIM - 1)];
    }
    rq[pos][t] = pq[pos][t] + pq[pos][t + 64];
    rk[pos][t] = pk[pos][t] + pk[pos][t + 64];
    __syncthreads();
    for (int s = 32; s > 0; s >>= 1) {
        if (t < s) { rq[pos][t] += rq[pos][t + s]; rk[pos][t] += rk[pos][t + s]; }
        __syncthreads();
    }
    const float invq = 1.f / rq[pos][0], invk = 1.f / rk[pos][0];
#pragma unroll
    for (int ii = 0; ii < 2; ++ii) {
        int i = t + ii * 64;
        float qi = pq[pos][i] * invq;
        float ki = pk[pos][i] * invk;
        knn[pos][i] = ki;
        qn[((size_t)bh * SEQ + n) * FDIM + i] = f2bf(qi);
        knT[(size_t)bh * FDIM * SEQ + (size_t)i * SEQ + n] = f2bf(ki);
    }
    __syncthreads();
    float vo = 0.f;
    const float* w0 = W0 + (size_t)bh * FDIM * DIM_HEAD;
    for (int s = 0; s < FDIM; ++s) vo += knn[pos][s] * w0[s * DIM_HEAD + t];
    const float dvv = beta[(size_t)bh * SEQ + n] * (vv - vo);
    dvT[(size_t)bh * DIM_HEAD * SEQ + (size_t)t * SEQ + n] = f2bf(dvv);  // transposed [d][n]
}

// W[bh] = W0[bh] + knT[bh] @ dv[bh]  (A=knT [128,4096], BT=dvT [64,4096]) -> WbfT [64][128]
__global__ void k_gemm_wupd(const __bf16* __restrict__ knT, const __bf16* __restrict__ dvT,
                            const float* __restrict__ W0, __bf16* __restrict__ WbfT) {
    const int bh = blockIdx.y;
    gemm_tile_bf16(knT + (size_t)bh * FDIM * SEQ, SEQ,
                   dvT + (size_t)bh * DIM_HEAD * SEQ, SEQ,
                   SEQ, 0, 0, 1.0f,
                   W0 + (size_t)bh * FDIM * DIM_HEAD, DIM_HEAD,
                   nullptr, nullptr,
                   nullptr, nullptr,
                   WbfT + (size_t)bh * FDIM * DIM_HEAD, FDIM, DIM_HEAD);
}

// outh[b,n,h*64+d] = SCALE * qn[bh] @ W[bh]  (A=qn [4096,128], BT=WbfT [64,128]) -> bf16
__global__ void k_gemm_apply(const __bf16* __restrict__ qn, const __bf16* __restrict__ WbfT,
                             __bf16* __restrict__ outh) {
    const int bh = blockIdx.y, b = bh >> 4, h = bh & 15;
    gemm_tile_bf16(qn + (size_t)bh * SEQ * FDIM, FDIM,
                   WbfT + (size_t)bh * FDIM * DIM_HEAD, FDIM,
                   FDIM, blockIdx.x * 128, 0, SCALE_F,
                   nullptr, 0, nullptr, nullptr,
                   nullptr, outh + (size_t)b * SEQ * DIM + h * DIM_HEAD, nullptr, 0, DIM);
}

// y = outh @ Wout + bout + x   [16384,1024]x[1024,1024] -> f32
__global__ void k_gemm_out(const __bf16* __restrict__ outh, const __bf16* __restrict__ woutT,
                           const float* __restrict__ bout, const float* __restrict__ x,
                           float* __restrict__ y) {
    const int ntiles = DIM / 64;               // 16
    int tm = blockIdx.x / ntiles, tn = blockIdx.x % ntiles;
    gemm_tile_bf16(outh, DIM, woutT, DIM, DIM, tm * 128, tn * 64, 1.0f,
                   nullptr, 0, bout, x, y, nullptr, nullptr, 0, DIM);
}

// Post-LayerNorm, one row (1024) per 256-thread block
__global__ void k_ln(const float* __restrict__ y, const float* __restrict__ gamma,
                     const float* __restrict__ betaln, float* __restrict__ out) {
    __shared__ float s1[256], s2[256];
    const int row = blockIdx.x, tid = threadIdx.x;
    const float* yr = y + (size_t)row * DIM;
    float a = 0.f, bsum = 0.f;
    for (int c = tid; c < DIM; c += 256) { float v = yr[c]; a += v; bsum += v * v; }
    s1[tid] = a; s2[tid] = bsum; __syncthreads();
    for (int s = 128; s > 0; s >>= 1) {
        if (tid < s) { s1[tid] += s1[tid + s]; s2[tid] += s2[tid + s]; }
        __syncthreads();
    }
    const float mean = s1[0] * (1.f / DIM);
    const float var  = s2[0] * (1.f / DIM) - mean * mean;
    const float inv  = rsqrtf(var + LN_EPS);
    for (int c = tid; c < DIM; c += 256)
        out[(size_t)row * DIM + c] = (yr[c] - mean) * inv * gamma[c] + betaln[c];
}

// ---------------------------------------------------------------------------
// Host launcher
// ---------------------------------------------------------------------------
extern "C" void kernel_launch(void* const* d_in, const int* in_sizes, int n_in,
                              void* d_out, int out_size, void* d_ws, size_t ws_size,
                              hipStream_t stream) {
    const float* x      = (const float*)d_in[0];
    const float* Wqkv   = (const float*)d_in[1];
    const float* Wbeta  = (const float*)d_in[2];
    const float* bbeta  = (const float*)d_in[3];
    const float* Wout   = (const float*)d_in[4];
    const float* bout   = (const float*)d_in[5];
    const float* gamma  = (const float*)d_in[6];
    const float* betaln = (const float*)d_in[7];
    const float* W0     = (const float*)d_in[8];
    float* out = (float*)d_out;

    char* w = (char*)d_ws;
    size_t off = 0;
    auto alloc = [&](size_t bytes) { char* p = w + off; off = (off + bytes + 255) & ~(size_t)255; return p; };

    __bf16* xbf    = (__bf16*)alloc((size_t)ROWS * DIM * 2);
    __bf16* wqkvT  = (__bf16*)alloc((size_t)3 * DIM * DIM * 2);   // [3072][1024]
    __bf16* woutT  = (__bf16*)alloc((size_t)DIM * DIM * 2);       // [1024][1024]
    float*  qkv    = (float*) alloc((size_t)ROWS * 3 * DIM * 4);
    float*  beta   = (float*) alloc((size_t)BH * SEQ * 4);
    __bf16* qn     = (__bf16*)alloc((size_t)BH * SEQ * FDIM * 2);
    __bf16* knT    = (__bf16*)alloc((size_t)BH * FDIM * SEQ * 2);
    __bf16* dvT    = (__bf16*)alloc((size_t)BH * DIM_HEAD * SEQ * 2);
    __bf16* WbfT   = (__bf16*)alloc((size_t)BH * FDIM * DIM_HEAD * 2);
    __bf16* outh   = (__bf16*)alloc((size_t)ROWS * DIM * 2);
    float*  ybuf   = (float*) alloc((size_t)ROWS * DIM * 4);
    (void)ws_size; (void)n_in; (void)in_sizes; (void)out_size;

    // 1. converts (weights transposed so every GEMM stages contiguous-K slabs)
    k_f32_to_bf16<<<2048, 256, 0, stream>>>(x, xbf, (long)ROWS * DIM);
    k_f32_to_bf16_T<<<dim3(3 * DIM / 32, DIM / 32), dim3(32, 8), 0, stream>>>(Wqkv, wqkvT, DIM, 3 * DIM);
    k_f32_to_bf16_T<<<dim3(DIM / 32, DIM / 32),     dim3(32, 8), 0, stream>>>(Wout, woutT, DIM, DIM);
    // 2. qkv projection (dominant GEMM)
    k_gemm_qkv<<<(ROWS / 128) * (3 * DIM / 64), 256, 0, stream>>>(xbf, wqkvT, qkv);
    // 3. beta
    k_beta<<<(ROWS * HEADS) / 256, 256, 0, stream>>>(x, Wbeta, bbeta, beta);
    // 4. dpfp features + dv (transposed)
    k_dpfp<<<(BH * SEQ) / 4, 256, 0, stream>>>(qkv, beta, W0, qn, knT, dvT);
    // 5. delta-rule weight update (batched over bh) -> W^T
    k_gemm_wupd<<<dim3(1, BH), 256, 0, stream>>>(knT, dvT, W0, WbfT);
    // 6. apply W to q (batched over bh)
    k_gemm_apply<<<dim3(SEQ / 128, BH), 256, 0, stream>>>(qn, WbfT, outh);
    // 7. output projection + bias + residual
    k_gemm_out<<<(ROWS / 128) * (DIM / 64), 256, 0, stream>>>(outh, woutT, bout, x, ybuf);
    // 8. LayerNorm
    k_ln<<<ROWS, 256, 0, stream>>>(ybuf, gamma, betaln, out);
}
